// EdgeConvBlock_44693429682215
// MI455X (gfx1250) — matrix operand order repaired and checked by reading, not compiled, not run on hardware
//
#include <hip/hip_runtime.h>
#include <hip/hip_bf16.h>
#include <math.h>

// ---------------------------------------------------------------------------
// EdgeConv block for MI455X (gfx1250, wave32, WMMA).
//   B=16, N=2048, D=64, C=128, K=16, GN groups of 4 channels.
// Compute-bound: ~43 GFLOP of GEMM work -> v_wmma_f32_16x16x32_f16.
// Memory traffic is ~25 MB -> irrelevant at 23.3 TB/s; all [B,N,K,C]
// intermediates are fused into registers/LDS. Weights staged into LDS via
// async global->LDS loads (ASYNCcnt) when the toolchain exposes them.
// ---------------------------------------------------------------------------

typedef __attribute__((ext_vector_type(16))) _Float16 v16h;
typedef __attribute__((ext_vector_type(8)))  _Float16 v8h;
typedef __attribute__((ext_vector_type(8)))  float    v8f;
typedef __attribute__((ext_vector_type(4)))  int      v4i;

#define BB 16
#define NN 2048
#define DD 64
#define CC 128
#define KNN 16
#define GN_EPS 1e-5f
#define INF_F 3.0e38f

#define AS_GLOBAL __attribute__((address_space(1)))
#define AS_LDS    __attribute__((address_space(3)))

#if defined(__AMDGCN__) && \
    __has_builtin(__builtin_amdgcn_global_load_async_to_lds_b128) && \
    __has_builtin(__builtin_amdgcn_s_wait_asynccnt)
#define HAVE_ASYNC_LDS 1
#endif

__device__ __forceinline__ v8f wmma_f16(v16h a, v16h b, v8f c) {
  return __builtin_amdgcn_wmma_f32_16x16x32_f16(
      /*neg_a=*/false, a, /*neg_b=*/false, b,
      /*c_mod=*/(short)0, c, /*reuse_a=*/false, /*reuse_b=*/false);
}

__device__ __forceinline__ v16h pack16(v8h lo, v8h hi) {
  v16h r;
#pragma unroll
  for (int i = 0; i < 8; ++i) { r[i] = lo[i]; r[8 + i] = hi[i]; }
  return r;
}

// Sorted-ascending top-16 insert (fully unrolled compare-swap chain,
// constant indices -> stays in VGPRs).
__device__ __forceinline__ void insert16(float (&tv)[16], int (&ti)[16],
                                         float v, int id) {
#pragma unroll
  for (int t = 0; t < 16; ++t) {
    bool sw = v < tv[t];
    float ov = tv[t]; int oi = ti[t];
    tv[t] = sw ? v : ov; ti[t] = sw ? id : oi;
    v    = sw ? ov : v;  id    = sw ? oi : id;
  }
}

// Cooperative 32 KB (16384-half) matrix copy global -> LDS, 256 threads.
// Uses GLOBAL_LOAD_ASYNC_TO_LDS_B128 (ASYNCcnt) when available.
__device__ __forceinline__ void stage_w(_Float16* dst /*LDS*/,
                                        const _Float16* src /*global*/) {
#if defined(HAVE_ASYNC_LDS)
#pragma unroll
  for (int r = 0; r < 8; ++r) {
    int id = r * 256 + (int)threadIdx.x;          // 2048 x 16B chunks
    __builtin_amdgcn_global_load_async_to_lds_b128(
        (AS_GLOBAL v4i*)((char*)src + (size_t)id * 16),
        (AS_LDS    v4i*)((char*)dst + (size_t)id * 16),
        /*offset=*/0, /*cpol=*/0);
  }
#else
#pragma unroll
  for (int r = 0; r < 8; ++r) {
    int id = r * 256 + (int)threadIdx.x;
    ((uint4*)dst)[id] = ((const uint4*)src)[id];
  }
#endif
}

__device__ __forceinline__ void stage_wait() {
#if defined(HAVE_ASYNC_LDS)
  __builtin_amdgcn_s_wait_asynccnt(0);
#endif
  __syncthreads();
}

// ---------------------------------------------------------------------------
// Prep: x -> f16 copy + squared norms
// ---------------------------------------------------------------------------
__global__ void prep_x_kernel(const float* __restrict__ x,
                              _Float16* __restrict__ xh,
                              float* __restrict__ x2) {
  int p = blockIdx.x * blockDim.x + threadIdx.x;
  if (p >= BB * NN) return;
  const float* xr = x + (size_t)p * DD;
  float s = 0.f;
#pragma unroll 8
  for (int d = 0; d < DD; ++d) {
    float v = xr[d];
    s += v * v;
    xh[(size_t)p * DD + d] = (_Float16)v;
  }
  x2[p] = s;
}

// W1[K=2D][C], W2[C][C] -> f16 transposed (Bt[n][k] = B[k][n]) so the
// WMMA B-fragment (lane = column, 16 contiguous K-halves per lane-half)
// is a single 32-byte load.
__global__ void prep_w_kernel(const float* __restrict__ W1,
                              const float* __restrict__ W2,
                              _Float16* __restrict__ w1t,
                              _Float16* __restrict__ w2t) {
  int t = blockIdx.x * blockDim.x + threadIdx.x;
  if (t >= CC * CC) return;
  int n = t >> 7, k = t & 127;       // t = n*128 + k
  w1t[t] = (_Float16)W1[(size_t)k * CC + n];
  w2t[t] = (_Float16)W2[(size_t)k * CC + n];
}

// ---------------------------------------------------------------------------
// kNN: one wave per 16-query tile. Gram via 2x WMMA(16x16x32) per 16-cand
// tile; distances bounced through a per-wave private LDS slot (LDS is
// in-order within a wave -> no barrier needed); each lane keeps a sorted
// top-16 over its 8 columns per tile; lane-halves merged with shfl_xor(16).
// ---------------------------------------------------------------------------
__global__ void __launch_bounds__(256)
knn_kernel(const _Float16* __restrict__ xh, const float* __restrict__ x2,
           const unsigned char* __restrict__ mask, int* __restrict__ idxout) {
  __shared__ float tile[8][16][17];   // +1 pad vs bank conflicts

  const int lane  = threadIdx.x & 31;
  const int wv    = threadIdx.x >> 5;
  const int gw    = blockIdx.x * 8 + wv;          // 2048 tiles total
  const int b     = gw / (NN / 16);
  const int i0    = (gw % (NN / 16)) * 16;
  const int row   = lane & 15;
  const int khalf = lane >> 4;

  // A fragments: 16 query rows x 64 K, split into two K=32 fragments.
  // A order used here (contiguous, same as B): lane holds K {kb..kb+15} with kb=khalf*16.
  const _Float16* qrow = xh + ((size_t)b * NN + i0 + row) * DD;
  const int kb = khalf * 16;
  v16h a0 = pack16(*(const v8h*)(qrow + kb),      *(const v8h*)(qrow + kb + 8));
  v16h a1 = pack16(*(const v8h*)(qrow + 32 + kb), *(const v8h*)(qrow + 32 + kb + 8));

  float x2q[8];
#pragma unroll
  for (int r = 0; r < 8; ++r) x2q[r] = x2[(size_t)b * NN + i0 + r + 8 * khalf];

  float topv[16]; int topi[16];
#pragma unroll
  for (int t = 0; t < 16; ++t) { topv[t] = INF_F; topi[t] = 0; }

  for (int jt = 0; jt < NN / 16; ++jt) {
    const int j0 = jt * 16;
    const int jc = j0 + (lane & 15);
    const _Float16* crow = xh + ((size_t)b * NN + jc) * DD;
    // B fragments: lane = column, 16 contiguous K per lane-half.
    v16h b0 = *(const v16h*)(crow + khalf * 16);
    v16h b1 = *(const v16h*)(crow + 32 + khalf * 16);

    v8f c = {};
    c = wmma_f16(a0, b0, c);
    c = wmma_f16(a1, b1, c);

    const float x2c = x2[(size_t)b * NN + jc];
    const bool  mj  = mask[(size_t)b * NN + jc] != 0;
#pragma unroll
    for (int r = 0; r < 8; ++r) {
      int M  = r + 8 * khalf;           // C layout: row M=r(+8), col=lane&15
      int gi = i0 + M;
      float d2 = fmaxf(x2q[r] + x2c - 2.f * c[r], 0.f);
      if (gi == jc || !mj) d2 = INF_F;
      tile[wv][M][lane & 15] = d2;
    }
    // wave-private LDS slot: intra-wave LDS ordering (DScnt) suffices,
    // no workgroup barrier.

    // lane owns row (lane&15), columns khalf*8 .. +8
#pragma unroll
    for (int t = 0; t < 8; ++t) {
      int col = khalf * 8 + t;
      insert16(topv, topi, tile[wv][lane & 15][col], j0 + col);
    }
  }

  // merge the two half-lane lists (both halves compute identical result)
  float pv[16]; int pi[16];
#pragma unroll
  for (int t = 0; t < 16; ++t) {
    pv[t] = __shfl_xor(topv[t], 16);
    pi[t] = __shfl_xor(topi[t], 16);
  }
#pragma unroll
  for (int t = 0; t < 16; ++t) insert16(topv, topi, pv[t], pi[t]);

  if (lane < 16) {
#pragma unroll
    for (int t = 0; t < 16; ++t)
      idxout[((size_t)b * NN + i0 + lane) * KNN + t] = topi[t];
  }
}

// ---------------------------------------------------------------------------
// Fused MLP: one wave per point.
//   E(16x128) = [x_nb - x_c | x_c]     (built in-register from f16 chunks)
//   H  = relu(GN(E*W1 + b1))           (GN via shfl_xor 1/2, 4-lane groups)
//   H2 = H*W2                          (H re-laid-out through LDS Hbuf)
//   out = max_k(H2) + b2, masked.
// W1^T then W2^T are ping-ponged through one 32 KB LDS buffer with async
// global->LDS loads; B-fragments become 2x ds_load_b128 per WMMA.
// ---------------------------------------------------------------------------
__global__ void __launch_bounds__(256)
mlp_kernel(const _Float16* __restrict__ xh, const int* __restrict__ idx,
           const float* __restrict__ b1, const float* __restrict__ gamma,
           const float* __restrict__ beta, const _Float16* __restrict__ w1t,
           const _Float16* __restrict__ w2t, const float* __restrict__ b2,
           const unsigned char* __restrict__ mask, float* __restrict__ out) {
  __shared__ _Float16 Wlds[CC * CC];      // 32 KB: active weight matrix
  __shared__ _Float16 Hbuf[8][16 * CC];   // 32 KB: per-wave 16x128 f16 H tile

  const int lane  = threadIdx.x & 31;
  const int wv    = threadIdx.x >> 5;
  const int p     = blockIdx.x * 8 + wv;  // point id, p = b*N + n
  const int col   = lane & 15;            // doubles as edge-row k for A frags
  const int khalf = lane >> 4;

  stage_w(Wlds, w1t);                     // async W1^T -> LDS

  const int nk = idx[(size_t)p * KNN + col];
  const int b  = p / NN;
  const _Float16* nrow = xh + ((size_t)b * NN + nk) * DD;  // neighbor row
  const _Float16* crow = xh + (size_t)p * DD;              // center row
  __builtin_prefetch(nrow, 0, 0);
  __builtin_prefetch(crow, 0, 0);

  stage_wait();                           // s_wait_asynccnt 0 + barrier

  // ---------------- GEMM1 + bias + GroupNorm + ReLU -> LDS ----------------
#pragma unroll
  for (int nt = 0; nt < 8; ++nt) {
    v8f acc = {};
#pragma unroll
    for (int ks = 0; ks < 4; ++ks) {
      // A fragment chunks: K in [K0,K0+8) and [K0+8,K0+16), K0=ks*32+khalf*16
      v8h lo, hi;
      if (ks < 2) {                      // edge cols 0..63: x_nb - x_c
        const int base = ks * 32 + khalf * 16;
        lo = *(const v8h*)(nrow + base)      - *(const v8h*)(crow + base);
        hi = *(const v8h*)(nrow + base + 8) - *(const v8h*)(crow + base + 8);
      } else {                           // edge cols 64..127: x_c
        const int base = (ks - 2) * 32 + khalf * 16;
        lo = *(const v8h*)(crow + base);
        hi = *(const v8h*)(crow + base + 8);
      }
      v16h a  = pack16(lo, hi);
      v16h bf = *(const v16h*)(&Wlds[(nt * 16 + col) * CC +
                                     ks * 32 + khalf * 16]);
      acc = wmma_f16(a, bf, acc);
    }

    const float b1c = b1[nt * 16 + col];
    const float gam = gamma[nt * 16 + col];
    const float bet = beta[nt * 16 + col];
#pragma unroll
    for (int r = 0; r < 8; ++r) {
      float h = acc[r] + b1c;
      // GN group = 4 consecutive channels = 4 adjacent lanes, same row.
      float s = h + __shfl_xor(h, 1);  s  += __shfl_xor(s, 2);
      float q = h * h;
      float sq = q + __shfl_xor(q, 1); sq += __shfl_xor(sq, 2);
      float mu  = s * 0.25f;
      float var = sq * 0.25f - mu * mu;
      float g = (h - mu) * rsqrtf(var + GN_EPS) * gam + bet;
      g = fmaxf(g, 0.f);
      Hbuf[wv][(r + 8 * khalf) * CC + nt * 16 + col] = (_Float16)g;
    }
  }
  __syncthreads();                        // all waves done reading W1^T

  stage_w(Wlds, w2t);                     // async W2^T -> LDS (overwrite)
  stage_wait();

  // ---------------- GEMM2 + K-max + bias + mask ----------------
  const float mk = mask[p] ? 1.f : 0.f;
#pragma unroll
  for (int nt = 0; nt < 8; ++nt) {
    v8f acc = {};
#pragma unroll
    for (int ks = 0; ks < 4; ++ks) {
      const _Float16* hrow = &Hbuf[wv][col * CC + ks * 32 + khalf * 16];
      v16h a  = pack16(*(const v8h*)hrow, *(const v8h*)(hrow + 8));
      v16h bf = *(const v16h*)(&Wlds[(nt * 16 + col) * CC +
                                     ks * 32 + khalf * 16]);
      acc = wmma_f16(a, bf, acc);
    }
    // column max over 16 edge rows: 8 rows in-register + partner half-lane
    float m = acc[0];
#pragma unroll
    for (int r = 1; r < 8; ++r) m = fmaxf(m, acc[r]);
    m = fmaxf(m, __shfl_xor(m, 16));
    m = (m + b2[nt * 16 + col]) * mk;
    if (lane < 16) out[(size_t)p * CC + nt * 16 + lane] = m;
  }
}

// ---------------------------------------------------------------------------
extern "C" void kernel_launch(void* const* d_in, const int* in_sizes, int n_in,
                              void* d_out, int out_size, void* d_ws, size_t ws_size,
                              hipStream_t stream) {
  const float*         x     = (const float*)d_in[0];
  const unsigned char* mask  = (const unsigned char*)d_in[1];  // bool array
  const float*         W1    = (const float*)d_in[2];
  const float*         b1    = (const float*)d_in[3];
  const float*         gam   = (const float*)d_in[4];
  const float*         bet   = (const float*)d_in[5];
  const float*         W2    = (const float*)d_in[6];
  const float*         b2    = (const float*)d_in[7];
  float*               out   = (float*)d_out;

  char* ws = (char*)d_ws;
  _Float16* xh  = (_Float16*)(ws);                         // 4 MB
  float*    x2  = (float*)(ws + 4194304);                  // 128 KB
  _Float16* w1t = (_Float16*)(ws + 4325376);               // 32 KB
  _Float16* w2t = (_Float16*)(ws + 4358144);               // 32 KB
  int*      idx = (int*)(ws + 4390912);                    // 2 MB

  prep_x_kernel<<<(BB * NN + 255) / 256, 256, 0, stream>>>(x, xh, x2);
  prep_w_kernel<<<(CC * CC + 255) / 256, 256, 0, stream>>>(W1, W2, w1t, w2t);
  // 2048 query tiles, 8 waves/block
  knn_kernel<<<256, 256, 0, stream>>>(xh, x2, mask, idx);
  // 32768 points, 8 waves/block
  mlp_kernel<<<BB * NN / 8, 256, 0, stream>>>(xh, idx, b1, gam, bet,
                                              w1t, w2t, b2, mask, out);
}
